// RecursiveNN_24068996726959
// MI455X (gfx1250) — matrix operand order repaired and checked by reading, not compiled, not run on hardware
//
#include <hip/hip_runtime.h>

#define B_TOT   2048
#define N_IN    64
#define N_OPS   1024
#define EMB     64
#define NNODES  (N_IN + N_OPS)   // 1088
#define TILE_B  16
#define THREADS 128              // 4 waves of 32

typedef __attribute__((ext_vector_type(16))) _Float16 v16h;
typedef __attribute__((ext_vector_type(8)))  _Float16 v8h;
typedef __attribute__((ext_vector_type(8)))  float    v8f;

union V16 {
    v16h v;
    v8h  p[2];
};

__global__ __launch_bounds__(THREADS, 1)
void recnn_wmma_kernel(const int*   __restrict__ input_ids,   // (B, N_IN)
                       const int*   __restrict__ op_w_idx,    // (N_OPS)
                       const int*   __restrict__ op_in1,      // (N_OPS)
                       const int*   __restrict__ op_in2,      // (N_OPS)
                       const float* __restrict__ embedding,   // (VOCAB, E)
                       const float* __restrict__ oio_W,       // (4, 64, 64)
                       const float* __restrict__ oio_b,       // (4, 64)
                       const float* __restrict__ tio_W,       // (4, 128, 64)
                       const float* __restrict__ tio_b,       // (4, 64)
                       const float* __restrict__ final_W,     // (64, 1)
                       const float* __restrict__ final_b,     // (1)
                       float*       __restrict__ out,         // (B, 1)
                       _Float16*    __restrict__ nodes)       // ws: [NNODES][B][E] f16
{
    // LDS: 8 unified weight matrices stored transposed as Wt[widx][e][k] (f16) + f32 biases.
    __shared__ _Float16 Wlds[8 * EMB * 128];     // 128 KB
    __shared__ float    blds[8 * EMB];           //   2 KB

    const int tid   = threadIdx.x;
    const int wave  = tid >> 5;        // 0..3  -> e-tile
    const int lane  = tid & 31;
    const int l15   = lane & 15;
    const int lhalf = lane >> 4;       // 0 or 1
    const int bBase = blockIdx.x * TILE_B;

    // ---- Phase 1: unified weights -> LDS (transposed; oio padded with zeros on k>=64)
    for (int idx = tid; idx < 8 * EMB * 128; idx += THREADS) {
        const int widx = idx >> 13;            // /(64*128)
        const int rem  = idx & 8191;
        const int e    = rem >> 7;             // /128
        const int k    = rem & 127;
        float w;
        if (widx < 4) {
            w = (k < 64) ? oio_W[(widx * 64 + k) * 64 + e] : 0.0f;
        } else {
            w = tio_W[((widx - 4) * 128 + k) * 64 + e];
        }
        Wlds[idx] = (_Float16)w;
    }
    for (int idx = tid; idx < 8 * EMB; idx += THREADS) {
        const int widx = idx >> 6;
        const int e    = idx & 63;
        blds[idx] = (widx < 4) ? oio_b[idx] : tio_b[(widx - 4) * 64 + e];
    }

    // ---- Phase 2: leaf embeddings -> f16 node slab (this WG's 16 batch rows only)
    for (int idx = tid; idx < TILE_B * N_IN * EMB; idx += THREADS) {
        const int b    = idx >> 12;            // /(64*64)
        const int rem  = idx & 4095;
        const int node = rem >> 6;
        const int e    = rem & 63;
        const int gb   = bBase + b;
        const int tok  = input_ids[gb * N_IN + node];
        nodes[((size_t)node * B_TOT + gb) * EMB + e] = (_Float16)embedding[tok * EMB + e];
    }
    __threadfence_block();
    __syncthreads();

    // ---- Phase 3: 1024 sequential ops, each a 64x128 @ 128x16 WMMA GEMM (transposed).
    // One barrier per step: B fragments are read directly from global (L0-resident),
    // A fragments from LDS, D written with a single 16B packed store per lane.
    const int ebase = wave * 16;

    int widx = op_w_idx[0];
    int i1   = op_in1[0];
    int i2   = op_in2[0];

    for (int j = 0; j < N_OPS; ++j) {
        // Prefetch next step's metadata (hidden behind this step's store + barrier).
        int nwidx = 0, ni1 = 0, ni2 = 0;
        if (j + 1 < N_OPS) {
            nwidx = op_w_idx[j + 1];
            ni1   = op_in1[j + 1];
            ni2   = op_in2[j + 1];
        }

        // Accumulator seeded with bias: D(e,b), e = ebase + r + 8*lhalf
        const float* bptr = blds + widx * EMB + ebase + lhalf * 8;
        v8f acc;
        #pragma unroll
        for (int r = 0; r < 8; ++r) acc[r] = bptr[r];

        // A lane base: Wt row for e = ebase + l15, K halves lhalf*8 + {0..7, 16..23}
        const _Float16* Arow = Wlds + ((size_t)(widx * EMB + ebase + l15)) * 128 + lhalf * 8;
        // B lane bases: x1/x2 row for b = bBase + l15, K halves lhalf*8 + {0..7, 16..23}
        const _Float16* B1 = nodes + ((size_t)i1 * B_TOT + bBase + l15) * EMB + lhalf * 8;
        const _Float16* B2 = nodes + ((size_t)i2 * B_TOT + bBase + l15) * EMB + lhalf * 8;

        #pragma unroll
        for (int c = 0; c < 4; ++c) {
            V16 a, bm;
            a.p[0] = *(const v8h*)(Arow + c * 32);
            a.p[1] = *(const v8h*)(Arow + c * 32 + 16);
            const _Float16* Bsrc = (c < 2) ? B1 : B2;
            const int eoff = (c & 1) * 32;          // chunk within x1 or x2
            bm.p[0] = *(const v8h*)(Bsrc + eoff);
            bm.p[1] = *(const v8h*)(Bsrc + eoff + 16);
            acc = __builtin_amdgcn_wmma_f32_16x16x32_f16(
                false, a.v, false, bm.v, (short)0, acc, false, false);
        }

        // ReLU via single-instruction clamp, pack to f16, one 16B store (e contiguous).
        union { v8h v; _Float16 h[8]; } o;
        #pragma unroll
        for (int r = 0; r < 8; ++r)
            o.h[r] = (_Float16)__builtin_amdgcn_fmed3f(acc[r], 0.0f, 3.0e38f);

        _Float16* dst = nodes + ((size_t)(N_IN + j) * B_TOT + bBase + l15) * EMB
                              + ebase + lhalf * 8;
        *(v8h*)dst = o.v;

        __threadfence_block();
        __syncthreads();

        widx = nwidx; i1 = ni1; i2 = ni2;
    }

    // ---- Phase 4: final Linear(E, 1) on the last node
    if (tid < TILE_B) {
        const int gb = bBase + tid;
        const _Float16* h = nodes + ((size_t)(NNODES - 1) * B_TOT + gb) * EMB;
        float s = final_b[0];
        #pragma unroll
        for (int e = 0; e < EMB; ++e)
            s += (float)h[e] * final_W[e];
        out[gb] = s;
    }
}

extern "C" void kernel_launch(void* const* d_in, const int* in_sizes, int n_in,
                              void* d_out, int out_size, void* d_ws, size_t ws_size,
                              hipStream_t stream) {
    const int*   input_ids = (const int*)  d_in[0];
    const int*   op_w_idx  = (const int*)  d_in[1];
    const int*   op_in1    = (const int*)  d_in[2];
    const int*   op_in2    = (const int*)  d_in[3];
    const float* embedding = (const float*)d_in[4];
    const float* oio_W     = (const float*)d_in[5];
    const float* oio_b     = (const float*)d_in[6];
    const float* tio_W     = (const float*)d_in[7];
    const float* tio_b     = (const float*)d_in[8];
    const float* final_W   = (const float*)d_in[9];
    const float* final_b   = (const float*)d_in[10];
    float*    out     = (float*)d_out;
    _Float16* scratch = (_Float16*)d_ws;   // needs NNODES*B*E*2 = ~272 MB

    recnn_wmma_kernel<<<B_TOT / TILE_B, THREADS, 0, stream>>>(
        input_ids, op_w_idx, op_in1, op_in2, embedding,
        oio_W, oio_b, tio_W, tio_b, final_W, final_b,
        out, scratch);
}
